// DeformableAttention_90701119357385
// MI455X (gfx1250) — matrix operand order repaired and checked by reading, not compiled, not run on hardware
//
#include <hip/hip_runtime.h>
#include <hip/hip_bf16.h>

// ---------------------------------------------------------------------------
// Deformable attention for MI455X (gfx1250, wave32, WMMA).
// Fixed problem size from the reference: B=4, H=W=128, N=16384, D=256,
// 8 heads x 32 dims, 4 points.
// ---------------------------------------------------------------------------

typedef _Float16 v16h __attribute__((ext_vector_type(16)));
typedef _Float16 v8h  __attribute__((ext_vector_type(8)));
typedef float    v8f  __attribute__((ext_vector_type(8)));

#define DA_B      4
#define DA_N      16384            // H*W
#define DA_M      (DA_B * DA_N)    // 65536 rows
#define DA_D      256
#define DA_K      256              // all GEMMs have K = 256

// ---------------------------------------------------------------------------
// f32 -> f16 elementwise convert (n multiple of 4)
// ---------------------------------------------------------------------------
__global__ __launch_bounds__(256) void cvt_f32_f16(const float* __restrict__ in,
                                                   _Float16* __restrict__ out,
                                                   int n) {
    int i = (blockIdx.x * 256 + threadIdx.x) * 4;
    if (i + 3 < n) {
        float4 v = *(const float4*)(in + i);
        out[i + 0] = (_Float16)v.x;
        out[i + 1] = (_Float16)v.y;
        out[i + 2] = (_Float16)v.z;
        out[i + 3] = (_Float16)v.w;
    }
}

// ---------------------------------------------------------------------------
// W [K, Nd] f32  ->  Wt [Nd, K] f16   (B-operand transposed, K contiguous)
// thread t = n*K + k  -> contiguous f16 writes
// ---------------------------------------------------------------------------
__global__ __launch_bounds__(256) void transpose_cvt(const float* __restrict__ W,
                                                     _Float16* __restrict__ Wt,
                                                     int K, int Nd) {
    int t = blockIdx.x * 256 + threadIdx.x;
    if (t >= K * Nd) return;
    int k = t % K;
    int n = t / K;
    Wt[(size_t)n * K + k] = (_Float16)W[(size_t)k * Nd + n];
}

// ---------------------------------------------------------------------------
// GEMM:  C[M,Nd] = A[M,K]_f16 * Bt[Nd,K]_f16^T + bias[Nd]
// 256 threads = 8 waves stacked in M; each wave owns a 16 x (NT*16) strip of
// output (NT accumulators), block tile = 128 x (NT*16).  K-loop in steps of
// 32 via V_WMMA_F32_16X16X32_F16: one A-fragment load (2 x global b128) is
// reused across NT WMMAs.  All NT B-fragment ds_loads for a k-step are issued
// BEFORE the WMMA chain so DScnt partial waits overlap LDS latency with
// matrix execution.
//
// Fragment layouts used here (16-bit A 16x32 / B 32x16 / f32 C; A/B use contiguous k, C per ISA):
//   A lane l: row = l%16; halves 0..15 = K[kbA .. kbA+15]  (kbA = l<16?0:16)
//   B lane l: col = l%16; halves 0..15 = K[kbB .. kbB+15]  (kbB = l<16?0:16)
//   C lane l, vgpr r: row = r + (l<16?0:8), col = l%16
// ---------------------------------------------------------------------------
template <int NT>
__global__ __launch_bounds__(256) void gemm_bias_wmma(
    const _Float16* __restrict__ A,    // [M, K] row-major f16
    const _Float16* __restrict__ Bt,   // [Nd, K] row-major f16 (B transposed)
    const float*    __restrict__ bias, // [Nd]
    void*           __restrict__ Cout, // [M, Nd] f32 or f16
    int M, int K, int Nd, int out_half) {

    __shared__ _Float16 __attribute__((aligned(64))) ldsB[64 * DA_K]; // 32 KB

    const int tid  = threadIdx.x;
    const int lane = tid & 31;
    const int wave = tid >> 5;
    const int bn   = NT * 16;                         // block tile N
    const int n0   = blockIdx.y * bn;                 // global col base
    const int m0   = blockIdx.x * 128 + wave * 16;    // 8 waves stacked in M

    // Stage Bt[n0 : n0+bn, 0:K] into LDS: straight contiguous copy.
    {
        const uint4* src = (const uint4*)(Bt + (size_t)n0 * K);
        uint4*       dst = (uint4*)ldsB;
        const int nvec = bn * K / 8;                  // 8 halves per uint4
        for (int i = tid; i < nvec; i += 256) dst[i] = src[i];
    }
    __syncthreads();

    const int row = m0 + (lane & 15);
    const int kbA = (lane < 16) ? 0 : 16;
    const int kbB = (lane < 16) ? 0 : 16;

    const _Float16* __restrict__ Arow = A + (size_t)row * K;
    const _Float16* __restrict__ Bl   = ldsB + (size_t)(lane & 15) * K;

    v8f acc[NT] = {};

    #pragma unroll 2
    for (int k0 = 0; k0 < K; k0 += 32) {
        __builtin_prefetch(Arow + k0 + 96, 0, 1);     // global_prefetch_b8
        v8h alo = *(const v8h*)(Arow + k0 + kbA);
        v8h ahi = *(const v8h*)(Arow + k0 + kbA + 8);
        v16h a  = __builtin_shufflevector(alo, ahi,
                    0, 1, 2, 3, 4, 5, 6, 7, 8, 9, 10, 11, 12, 13, 14, 15);
        // Issue ALL B-fragment loads up-front -> partial DScnt waits let the
        // LDS latency of tiles 1..NT-1 hide under WMMAs of tiles 0..NT-2.
        v16h b[NT];
        #pragma unroll
        for (int t = 0; t < NT; ++t)
            b[t] = *(const v16h*)(Bl + (size_t)t * 16 * K + k0 + kbB);
        #pragma unroll
        for (int t = 0; t < NT; ++t)
            acc[t] = __builtin_amdgcn_wmma_f32_16x16x32_f16(
                        /*neg_a=*/false, a, /*neg_b=*/false, b[t],
                        /*c_mod=*/(short)0, acc[t],
                        /*reuse_a=*/false, /*reuse_b=*/false);
    }

    const int rbase = m0 + ((lane >> 4) << 3);        // +8 for lanes 16..31

    #pragma unroll
    for (int t = 0; t < NT; ++t) {
        const int   colg = n0 + t * 16 + (lane & 15);
        const float bv   = bias[colg];
        if (out_half) {
            _Float16* O = (_Float16*)Cout;
            #pragma unroll
            for (int r = 0; r < 8; ++r)
                O[(size_t)(rbase + r) * Nd + colg] = (_Float16)(acc[t][r] + bv);
        } else {
            float* O = (float*)Cout;
            #pragma unroll
            for (int r = 0; r < 8; ++r)
                O[(size_t)(rbase + r) * Nd + colg] = acc[t][r] + bv;
        }
    }
}

// ---------------------------------------------------------------------------
// Bilinear multi-point sampling.  One wave32 per (b, n, head); lane = channel.
// Softmax over the 4 point-logits in registers; 16 coalesced 64B f16 gathers.
// ---------------------------------------------------------------------------
__device__ __forceinline__ float da_corner(const _Float16* __restrict__ vb,
                                           int yi, int xi, int h, int w,
                                           float wt) {
    bool valid = (xi >= 0) & (xi < w) & (yi >= 0) & (yi < h);
    int cy = yi < 0 ? 0 : (yi > h - 1 ? h - 1 : yi);
    int cx = xi < 0 ? 0 : (xi > w - 1 ? w - 1 : xi);
    float v = (float)vb[(size_t)(cy * w + cx) * DA_D];
    return valid ? wt * v : 0.0f;
}

__global__ __launch_bounds__(256) void msda_sample(
    const _Float16* __restrict__ Vh,     // [B*N, 256]  (b,n,head,dim)
    const float*    __restrict__ off,    // [B*N, 64]   (head,point,xy)
    const float*    __restrict__ logits, // [B*N, 32]   (head,point)
    const float*    __restrict__ ref,    // [B*N, 2]
    const int*      __restrict__ shapes, // [1,2] = {H, W}
    _Float16*       __restrict__ out) {  // [B*N, 256]

    const int lane = threadIdx.x & 31;
    const int wave = threadIdx.x >> 5;
    const int gid  = blockIdx.x * 8 + wave;   // (b*N + n)*8 + head
    const int head = gid & 7;
    const int q    = gid >> 3;                // b*N + n
    const int b    = q >> 14;                 // N = 16384

    const int h = shapes[0];
    const int w = shapes[1];

    const float rx = ref[q * 2 + 0];
    const float ry = ref[q * 2 + 1];
    const float* __restrict__ o  = off    + (size_t)q * 64 + head * 8;
    const float* __restrict__ lg = logits + (size_t)q * 32 + head * 4;

    // softmax over the 4 points
    float e[4];
    float l0 = lg[0], l1 = lg[1], l2 = lg[2], l3 = lg[3];
    float mx = fmaxf(fmaxf(l0, l1), fmaxf(l2, l3));
    e[0] = __expf(l0 - mx); e[1] = __expf(l1 - mx);
    e[2] = __expf(l2 - mx); e[3] = __expf(l3 - mx);
    float inv = 1.0f / (e[0] + e[1] + e[2] + e[3]);

    const _Float16* __restrict__ vb =
        Vh + ((size_t)b * DA_N) * DA_D + head * 32 + lane;

    float acc = 0.0f;
    #pragma unroll
    for (int p = 0; p < 4; ++p) {
        float aw = e[p] * inv;
        // loc = ref + off / (W, H);  pixel = loc * (W, H) - 0.5
        float x = (rx + o[2 * p + 0] / (float)w) * (float)w - 0.5f;
        float y = (ry + o[2 * p + 1] / (float)h) * (float)h - 0.5f;
        float fx0 = floorf(x), fy0 = floorf(y);
        int   x0 = (int)fx0,   y0 = (int)fy0;
        float lx = x - fx0,    ly = y - fy0;
        float s = da_corner(vb, y0,     x0,     h, w, (1.f - ly) * (1.f - lx))
                + da_corner(vb, y0,     x0 + 1, h, w, (1.f - ly) * lx)
                + da_corner(vb, y0 + 1, x0,     h, w, ly * (1.f - lx))
                + da_corner(vb, y0 + 1, x0 + 1, h, w, ly * lx);
        acc += aw * s;
    }
    out[(size_t)q * DA_D + head * 32 + lane] = (_Float16)acc;
}

// ---------------------------------------------------------------------------
// Host-side launch
// ---------------------------------------------------------------------------
extern "C" void kernel_launch(void* const* d_in, const int* in_sizes, int n_in,
                              void* d_out, int out_size, void* d_ws, size_t ws_size,
                              hipStream_t stream) {
    const float* query  = (const float*)d_in[0];   // [B,N,256]
    const float* refpts = (const float*)d_in[1];   // [B,N,2]
    const float* value  = (const float*)d_in[2];   // [B,N,256]
    const int*   shapes = (const int*)d_in[3];     // [1,2]
    /* d_in[4] level_start_index unused */
    const float* W_so = (const float*)d_in[5];     // [256, 64]
    const float* b_so = (const float*)d_in[6];     // [64]
    const float* W_aw = (const float*)d_in[7];     // [256, 32]
    const float* b_aw = (const float*)d_in[8];     // [32]
    const float* W_v  = (const float*)d_in[9];     // [256, 256]
    const float* b_v  = (const float*)d_in[10];    // [256]
    const float* W_o  = (const float*)d_in[11];    // [256, 256]
    const float* b_o  = (const float*)d_in[12];    // [256]

    // ---- workspace carve-up (all 256B aligned) ----
    char* ws = (char*)d_ws;
    _Float16* q_h  = (_Float16*)ws; ws += (size_t)DA_M * DA_D * 2;   // 32 MB
    _Float16* v_h  = (_Float16*)ws; ws += (size_t)DA_M * DA_D * 2;   // 32 MB
    _Float16* Vh   = (_Float16*)ws; ws += (size_t)DA_M * DA_D * 2;   // 32 MB
    _Float16* samp = (_Float16*)ws; ws += (size_t)DA_M * DA_D * 2;   // 32 MB
    float*    offb = (float*)ws;    ws += (size_t)DA_M * 64 * 4;     // 16 MB
    float*    awb  = (float*)ws;    ws += (size_t)DA_M * 32 * 4;     //  8 MB
    _Float16* Wv_t  = (_Float16*)ws; ws += (size_t)DA_K * 256 * 2;
    _Float16* Wo_t  = (_Float16*)ws; ws += (size_t)DA_K * 256 * 2;
    _Float16* Wso_t = (_Float16*)ws; ws += (size_t)DA_K * 64 * 2;
    _Float16* Waw_t = (_Float16*)ws; ws += (size_t)DA_K * 32 * 2;

    const int elems = DA_M * DA_D;                   // 16.7M

    // 1) precision conversion of the streaming operands
    cvt_f32_f16<<<elems / (4 * 256), 256, 0, stream>>>(query, q_h, elems);
    cvt_f32_f16<<<elems / (4 * 256), 256, 0, stream>>>(value, v_h, elems);

    // 2) transpose+convert weights -> [Nd, K] f16
    transpose_cvt<<<(DA_K * 256 + 255) / 256, 256, 0, stream>>>(W_v,  Wv_t,  DA_K, 256);
    transpose_cvt<<<(DA_K * 256 + 255) / 256, 256, 0, stream>>>(W_o,  Wo_t,  DA_K, 256);
    transpose_cvt<<<(DA_K * 64  + 255) / 256, 256, 0, stream>>>(W_so, Wso_t, DA_K, 64);
    transpose_cvt<<<(DA_K * 32  + 255) / 256, 256, 0, stream>>>(W_aw, Waw_t, DA_K, 32);

    // 3) V = value @ W_v + b_v   (f16 out, block tile 128x64, NT=4)
    gemm_bias_wmma<4><<<dim3(DA_M / 128, 256 / 64), 256, 0, stream>>>(
        v_h, Wv_t, b_v, (void*)Vh, DA_M, DA_K, 256, /*out_half=*/1);

    // 4) off = query @ W_so + b_so   (f32 out, block tile 128x64, NT=4)
    gemm_bias_wmma<4><<<dim3(DA_M / 128, 1), 256, 0, stream>>>(
        q_h, Wso_t, b_so, (void*)offb, DA_M, DA_K, 64, 0);

    // 5) logits = query @ W_aw + b_aw  (f32 out, block tile 128x32, NT=2)
    gemm_bias_wmma<2><<<dim3(DA_M / 128, 1), 256, 0, stream>>>(
        q_h, Waw_t, b_aw, (void*)awb, DA_M, DA_K, 32, 0);

    // 6) bilinear sampling + softmax  -> sampled (f16)
    msda_sample<<<DA_M * 8 / 8, 256, 0, stream>>>(
        Vh, offb, awb, refpts, shapes, samp);

    // 7) out = sampled @ W_o + b_o   (f32 -> d_out, NT=4)
    gemm_bias_wmma<4><<<dim3(DA_M / 128, 256 / 64), 256, 0, stream>>>(
        samp, Wo_t, b_o, d_out, DA_M, DA_K, 256, 0);

    (void)in_sizes; (void)n_in; (void)out_size; (void)ws_size;
}
